// LuongAttention_83296595739074
// MI455X (gfx1250) — matrix operand order repaired and checked by reading, not compiled, not run on hardware
//
#include <hip/hip_runtime.h>
#include <hip/hip_bf16.h>

// Problem constants (match reference)
#define BB 8
#define SS 4096
#define DD 1024
#define QS 1032   // padded LDS row stride in bf16 elems (mult of 8, stride%64dw=4 -> no bank conflicts)

typedef __attribute__((ext_vector_type(16))) __bf16 v16bf;
typedef __attribute__((ext_vector_type(8)))  float  v8f;

struct BF16x16 { uint4 lo; uint4 hi; };   // 32 bytes == v16bf

__device__ __forceinline__ unsigned short f2bf(float f) {
    unsigned int u = __float_as_uint(f);
    u += 0x7FFFu + ((u >> 16) & 1u);       // round-to-nearest-even
    return (unsigned short)(u >> 16);
}

// ---------------------------------------------------------------------------
// Kernel 0: W [k][n] f32  ->  Wt [n][k] bf16 (transposed, for contiguous-K B-fragments)
// ---------------------------------------------------------------------------
__global__ void prep_w_kernel(const float* __restrict__ w, unsigned short* __restrict__ wt) {
    __shared__ float tile[32][33];
    const int bx = blockIdx.x;   // k tile
    const int by = blockIdx.y;   // n tile
    const int tx = threadIdx.x;  // 0..31
    const int ty = threadIdx.y;  // 0..7
#pragma unroll
    for (int r = 0; r < 4; ++r) {
        int k = bx * 32 + ty + r * 8;
        int n = by * 32 + tx;
        tile[ty + r * 8][tx] = w[(size_t)k * DD + n];
    }
    __syncthreads();
#pragma unroll
    for (int r = 0; r < 4; ++r) {
        int n = by * 32 + ty + r * 8;
        int k = bx * 32 + tx;
        wt[(size_t)n * DD + k] = f2bf(tile[tx][ty + r * 8]);
    }
}

// ---------------------------------------------------------------------------
// Kernel 1: logits  w[row] = q[row,:] @ W @ k[row,:]   for row in [0, B*S)
// One wave (32 threads) per strip of 16 rows. Fused GEMM + rowwise dot:
// acc += (sum_k A*B) .* Kvals per 16-column N tile; dq never hits memory.
// ---------------------------------------------------------------------------
__global__ void __launch_bounds__(32)
phase1_kernel(const float* __restrict__ q,            // [B*S, D] f32
              const float* __restrict__ kmat,         // [B*S, D] f32
              const unsigned short* __restrict__ wt,  // [D][D] bf16 bits, W^T
              float* __restrict__ wout)               // [B*S] f32
{
    __shared__ unsigned short qs[16 * QS];            // 33 KB
    const int  lane    = threadIdx.x;
    const long rowbase = (long)blockIdx.x * 16;

    // ---- stage 16x1024 Q strip into LDS as bf16 (coalesced float4 reads) ----
    for (int i = lane; i < 16 * 256; i += 32) {
        const int r  = i >> 8;
        const int c4 = i & 255;
        float4 f = ((const float4*)(q + (rowbase + r) * DD))[c4];
        uint2 p;
        p.x = ((unsigned)f2bf(f.y) << 16) | f2bf(f.x);
        p.y = ((unsigned)f2bf(f.w) << 16) | f2bf(f.z);
        *(uint2*)&qs[r * QS + c4 * 4] = p;
    }
    __syncthreads();

    const int half = lane >> 4;     // 0 | 1
    const int m16  = lane & 15;

    v8f acc = {};
    for (int nt = 0; nt < 64; ++nt) {
        const int nbase = nt * 16;
        v8f c = {};
#pragma unroll 4
        for (int kt = 0; kt < 32; ++kt) {
            const int kb = kt * 32;
            // A fragment: 16x32 bf16, lane m16 = row; 16 contiguous K
            const unsigned short* ap = &qs[m16 * QS + kb + half * 16];
            BF16x16 ab;
            ab.lo = *(const uint4*)(ap);        // K = kb + half*16 .. +8
            ab.hi = *(const uint4*)(ap + 8);    // K = kb + half*16 + 8 .. +8
            // B fragment: 32x16 bf16, lane m16 = col N; 16 contiguous K from Wt
            const unsigned short* bp = wt + (size_t)(nbase + m16) * DD + kb + half * 16;
            BF16x16 bb;
            bb.lo = *(const uint4*)(bp);
            bb.hi = *(const uint4*)(bp + 8);
            v16bf a = __builtin_bit_cast(v16bf, ab);
            v16bf b = __builtin_bit_cast(v16bf, bb);
            c = __builtin_amdgcn_wmma_f32_16x16x32_bf16(
                    false, a, false, b, (short)0, c, false, false);
        }
        // elementwise multiply by K values in C layout, accumulate over N tiles
        const float* kp = kmat + (rowbase + half * 8) * DD + nbase + m16;
#pragma unroll
        for (int j = 0; j < 8; ++j)
            acc[j] += c[j] * kp[(size_t)j * DD];
    }

    // reduce over the 16 lanes of each half (columns of the tile)
#pragma unroll
    for (int j = 0; j < 8; ++j) {
        float v = acc[j];
        v += __shfl_xor(v, 1, 32);
        v += __shfl_xor(v, 2, 32);
        v += __shfl_xor(v, 4, 32);
        v += __shfl_xor(v, 8, 32);
        acc[j] = v;
    }
    if (m16 == 0) {
#pragma unroll
        for (int j = 0; j < 8; ++j)
            wout[rowbase + half * 8 + j] = acc[j];
    }
}

// ---------------------------------------------------------------------------
// Kernel 2: masked softmax over S per batch (in-place logits -> coef)
// ---------------------------------------------------------------------------
__global__ void softmax_kernel(const int* __restrict__ seqlen, float* __restrict__ w) {
    const int b   = blockIdx.x;
    const int tid = threadIdx.x;          // 256 threads
    const int len = seqlen[b];
    float* wb = w + (size_t)b * SS;

    float vals[16];
    float mx = -3.402823466e38f;
#pragma unroll
    for (int i = 0; i < 16; ++i) {
        const int s = tid + i * 256;
        const float v = (s < len) ? wb[s] : -3.402823466e38f;
        vals[i] = v;
        mx = fmaxf(mx, v);
    }
    __shared__ float red[256];
    red[tid] = mx;
    __syncthreads();
    for (int off = 128; off > 0; off >>= 1) {
        if (tid < off) red[tid] = fmaxf(red[tid], red[tid + off]);
        __syncthreads();
    }
    mx = red[0];
    __syncthreads();

    float sum = 0.f;
#pragma unroll
    for (int i = 0; i < 16; ++i) {
        const int s = tid + i * 256;
        const float e = (s < len) ? __expf(vals[i] - mx) : 0.f;
        vals[i] = e;
        sum += e;
    }
    red[tid] = sum;
    __syncthreads();
    for (int off = 128; off > 0; off >>= 1) {
        if (tid < off) red[tid] += red[tid + off];
        __syncthreads();
    }
    const float inv = 1.f / red[0];
#pragma unroll
    for (int i = 0; i < 16; ++i)
        wb[tid + i * 256] = vals[i] * inv;
}

// ---------------------------------------------------------------------------
// Kernel 3: out[b,v] = sum_s coef[b,s] * V[b,s,v]   (S-chunked, atomic combine)
// ---------------------------------------------------------------------------
__global__ void zero_out_kernel(float* __restrict__ out) {
    const int i = blockIdx.x * 256 + threadIdx.x;
    if (i < BB * DD) out[i] = 0.f;
}

__global__ void wsum_kernel(const float* __restrict__ vmat,
                            const float* __restrict__ coef,
                            float* __restrict__ out) {
    const int v  = blockIdx.x * 256 + threadIdx.x;  // gridDim.x = D/256
    const int b  = blockIdx.z;
    const int s0 = blockIdx.y * 512;                // gridDim.y = S/512
    const float* vb = vmat + ((size_t)b * SS + s0) * DD + v;
    const float* cb = coef + (size_t)b * SS + s0;
    float acc = 0.f;
#pragma unroll 4
    for (int s = 0; s < 512; ++s)
        acc += cb[s] * vb[(size_t)s * DD];
    atomicAdd(&out[b * DD + v], acc);
}

// ---------------------------------------------------------------------------
extern "C" void kernel_launch(void* const* d_in, const int* in_sizes, int n_in,
                              void* d_out, int out_size, void* d_ws, size_t ws_size,
                              hipStream_t stream) {
    const float* keys    = (const float*)d_in[0];
    const float* queries = (const float*)d_in[1];
    const float* values  = (const float*)d_in[2];
    const float* W       = (const float*)d_in[3];
    const int*   seqlen  = (const int*)  d_in[4];
    float* out = (float*)d_out;

    // workspace: [0, 2MiB) Wt bf16 ; [2MiB, 2MiB+128KiB) logits/coef f32
    unsigned short* wt  = (unsigned short*)d_ws;
    float*          wbuf = (float*)((char*)d_ws + (size_t)DD * DD * sizeof(unsigned short));

    prep_w_kernel<<<dim3(DD / 32, DD / 32), dim3(32, 8), 0, stream>>>(W, wt);
    phase1_kernel<<<(BB * SS) / 16, 32, 0, stream>>>(queries, keys, wt, wbuf);
    softmax_kernel<<<BB, 256, 0, stream>>>(seqlen, wbuf);
    zero_out_kernel<<<(BB * DD + 255) / 256, 256, 0, stream>>>(out);
    wsum_kernel<<<dim3(DD / 256, SS / 512, BB), 256, 0, stream>>>(values, wbuf, out);
}
